// _DSPPModule_14345190768896
// MI455X (gfx1250) — matrix operand order repaired and checked by reading, not compiled, not run on hardware
//
#include <hip/hip_runtime.h>
#include <hip/hip_bf16.h>
#include <stdint.h>

// ---------------- problem constants (match reference) ----------------
#define NPT   9         // KS*KS sampling points
#define DILV  6
#define PADV  6
#define BATCH 4
#define CIN   256
#define HH    64
#define WW    64
#define HWSZ  (HH*WW)       // 4096
#define POUT  256
#define KTOT  (CIN*NPT)     // 2304
#define NCHUNK (KTOT/32)    // 72
#define HP    (HH + 2*PADV) // 76 (padded extent)
#define EPSV  1e-5f

typedef __attribute__((ext_vector_type(16))) __bf16 v16bf;
typedef __attribute__((ext_vector_type(8)))  __bf16 v8bf;
typedef __attribute__((ext_vector_type(8)))  float  v8f;
typedef __attribute__((ext_vector_type(4)))  unsigned int u32x4;
typedef __attribute__((ext_vector_type(8)))  int i32x8;
typedef __attribute__((ext_vector_type(4)))  int i32x4;

// ---------------------------------------------------------------------
// Kernel 0: transpose+convert main weight w[P][C][3][3] (f32)
//           -> wT[p][k] bf16 with k = n*256 + c  (K contiguous per column)
// ---------------------------------------------------------------------
__global__ void k_convert_w(const float* __restrict__ w, __bf16* __restrict__ wT) {
    int idx = blockIdx.x * 256 + threadIdx.x;       // over POUT*KTOT
    if (idx >= POUT * KTOT) return;
    int p = idx / KTOT, k = idx - p * KTOT;
    int n = k >> 8, c = k & 255;
    wT[idx] = (__bf16)w[((size_t)p * CIN + c) * NPT + n];
}

// ---------------------------------------------------------------------
// Kernel 1: offset (18 ch) + modulation mask (9 ch, sigmoid) 3x3 conv,
//           padding=1. One block per pixel; thread = input channel;
//           wave32 shfl reduction then cross-wave LDS reduction.
//           om layout: [b][27][HW]  (j<9: off_x, j<18: off_y, j>=18: m)
// ---------------------------------------------------------------------
__global__ void __launch_bounds__(256)
k_offset_mask(const float* __restrict__ x,
              const float* __restrict__ w_p, const float* __restrict__ b_p,
              const float* __restrict__ w_m, const float* __restrict__ b_m,
              float* __restrict__ om) {
    int pix = blockIdx.x;                 // 0..16383
    int b = pix >> 12, hw = pix & 4095;
    int h = hw >> 6, wc = hw & 63;
    int c = threadIdx.x;
    int lane = threadIdx.x & 31, wv = threadIdx.x >> 5;

    const float* xp = x + ((size_t)(b * CIN + c)) * HWSZ;
    float xv[9];
#pragma unroll
    for (int kh = 0; kh < 3; ++kh)
#pragma unroll
        for (int kw = 0; kw < 3; ++kw) {
            int hh = h + kh - 1, ww2 = wc + kw - 1;
            xv[kh * 3 + kw] = (hh >= 0 && hh < HH && ww2 >= 0 && ww2 < WW)
                              ? xp[hh * WW + ww2] : 0.f;
        }

    __shared__ float red[27][8];
    for (int j = 0; j < 27; ++j) {
        const float* wj = (j < 18) ? (w_p + ((size_t)(j * CIN + c)) * 9)
                                   : (w_m + ((size_t)((j - 18) * CIN + c)) * 9);
        float s = 0.f;
#pragma unroll
        for (int t = 0; t < 9; ++t) s += wj[t] * xv[t];
#pragma unroll
        for (int m = 16; m >= 1; m >>= 1) s += __shfl_xor(s, m, 32);
        if (lane == 0) red[j][wv] = s;
    }
    __syncthreads();
    if (threadIdx.x < 27) {
        int j = threadIdx.x;
        float s = 0.f;
#pragma unroll
        for (int wvi = 0; wvi < 8; ++wvi) s += red[j][wvi];
        if (j < 18) s += b_p[j];
        else { s += b_m[j - 18]; s = 1.f / (1.f + __expf(-s)); }
        om[((size_t)(b * 27 + j)) * HWSZ + hw] = s;
    }
}

// ---------------------------------------------------------------------
// Kernel 2: fused deformable sampling + WMMA GEMM, double-buffered.
//   Block = 256 threads (8 waves), M-tile = 16 pixels, all 256 out-cols.
//   K loop: 72 chunks of 32 (k = n*256 + c). Software pipeline:
//     chunk i+1: wave0 issues TDM tensor_load_to_lds of B tile -> lds_B[nxt],
//                all threads build bilinear-sampled A tile -> lds_A[nxt]
//     chunk i  : 2x v_wmma_f32_16x16x32_bf16 per wave on lds_A/B[cur]
//   One s_wait_tensorcnt + one barrier per chunk.
//   Gathers are branchless: OOB corners carry offset 0 / weight 0.
// ---------------------------------------------------------------------
__device__ __forceinline__ void tdm_load_B(const __bf16* gsrc, unsigned lds_addr) {
    unsigned long long ga = (unsigned long long)(uintptr_t)gsrc;
    u32x4 g0 = { 1u,                                       // count=1
                 lds_addr,                                 // lds_addr
                 (unsigned)(ga & 0xFFFFFFFFu),             // global_addr lo
                 (unsigned)((ga >> 32) & 0x01FFFFFFu) | (2u << 30) }; // hi + type=2
    i32x8 g1 = { (int)(1u << 16),                          // data_size = 2B
                 (int)((KTOT & 0xFFFF) << 16),             // tensor_dim0 lo
                 (int)(POUT << 16) | (KTOT >> 16),         // dim0 hi | dim1 lo
                 (int)(32 << 16),                          // tile_dim0 = 32
                 POUT,                                     // tile_dim1 = 256
                 KTOT,                                     // tensor_dim0_stride
                 0, 0 };
    i32x4 gz = { 0, 0, 0, 0 };
#if defined(__clang_major__) && (__clang_major__ >= 23)
    i32x8 gz8 = { 0, 0, 0, 0, 0, 0, 0, 0 };
    __builtin_amdgcn_tensor_load_to_lds(g0, g1, gz, gz, gz8, 0);
#else
    __builtin_amdgcn_tensor_load_to_lds(g0, g1, gz, gz, 0);
#endif
}

__global__ void __launch_bounds__(256)
k_deform_wmma(const float* __restrict__ x, const float* __restrict__ om,
              const __bf16* __restrict__ wT, float* __restrict__ y) {
    __shared__ __align__(16) __bf16 lds_B[2][POUT][32];  // 32 KB, [buf][col p][k]
    __shared__ __align__(16) __bf16 lds_A[2][16][32];    //  2 KB, [buf][pixel][k]
    __shared__ __align__(16) int   s_ofs[16][NPT][4];    // gather plane offsets (0 if OOB)
    __shared__ __align__(16) float s_wgt[16][NPT][4];    // bilinear wgt * mask (0 if OOB)

    const int pix0 = blockIdx.x * 16;
    const int b = pix0 >> 12;
    const int hw0 = pix0 & 4095;
    const int tid = threadIdx.x, lane = tid & 31, wv = tid >> 5;

    // ---- precompute sampling data: 16 pixels x 9 points ----
    if (tid < 16 * NPT) {
        int p = tid / NPT, n = tid - p * NPT;
        int hw = hw0 + p;
        int h = hw >> 6, wc = hw & 63;
        float offx = om[((size_t)(b * 27 +      n)) * HWSZ + hw];
        float offy = om[((size_t)(b * 27 +  9 + n)) * HWSZ + hw];
        float mval = om[((size_t)(b * 27 + 18 + n)) * HWSZ + hw];
        float px = (float)(h + 1)  + (float)((n / 3 - 1) * DILV) + offx;
        float py = (float)(wc + 1) + (float)((n % 3 - 1) * DILV) + offy;
        float fx = floorf(px), fy = floorf(py);
        float qltx = fminf(fmaxf(fx,       0.f), (float)(HP - 1));
        float qlty = fminf(fmaxf(fy,       0.f), (float)(HP - 1));
        float qrbx = fminf(fmaxf(fx + 1.f, 0.f), (float)(HP - 1));
        float qrby = fminf(fmaxf(fy + 1.f, 0.f), (float)(HP - 1));
        px = fminf(fmaxf(px, 0.f), (float)(HP - 1));
        py = fminf(fmaxf(py, 0.f), (float)(HP - 1));
        float glt = (1.f + (qltx - px)) * (1.f + (qlty - py));
        float grb = (1.f - (qrbx - px)) * (1.f - (qrby - py));
        float glb = (1.f + (qltx - px)) * (1.f - (qrby - py));
        float grt = (1.f - (qrbx - px)) * (1.f + (qlty - py));
        int iltx = (int)qltx, ilty = (int)qlty, irbx = (int)qrbx, irby = (int)qrby;
        // branchless OOB: offset 0, weight 0 (pad ring of x_pad is zero anyway)
        auto emit = [&](int slot, int qx, int qy, float g) {
            int rx = qx - PADV, ry = qy - PADV;      // padded -> real coords
            bool ok = (rx >= 0 && rx < HH && ry >= 0 && ry < WW);
            s_ofs[p][n][slot] = ok ? (rx * WW + ry) : 0;
            s_wgt[p][n][slot] = ok ? g * mval : 0.f;
        };
        emit(0, iltx, ilty, glt);
        emit(1, irbx, irby, grb);
        emit(2, iltx, irby, glb);
        emit(3, irbx, ilty, grt);
    }

    const float* xb = x + ((size_t)b) * CIN * HWSZ;
    const unsigned addrB0 = (unsigned)(uintptr_t)&lds_B[0][0][0];
    const unsigned addrB1 = (unsigned)(uintptr_t)&lds_B[1][0][0];
    const int p_a  = tid >> 4;          // pixel this thread samples for
    const int cc_a = (tid & 15) * 2;    // 2 adjacent channels per thread

    __syncthreads();   // sampling tables ready

    // branchless A-tile builder for chunk kc into buffer buf
    auto build_A = [&](int kc, int buf) {
        int n  = kc >> 3;
        int c0 = (kc & 7) * 32;
        int4   of = *(const int4*)  &s_ofs[p_a][n][0];
        float4 wg = *(const float4*)&s_wgt[p_a][n][0];
        const float* xp0 = xb + (size_t)(c0 + cc_a) * HWSZ;
        const float* xp1 = xp0 + HWSZ;
        float a0 = wg.x * xp0[of.x] + wg.y * xp0[of.y]
                 + wg.z * xp0[of.z] + wg.w * xp0[of.w];
        float a1 = wg.x * xp1[of.x] + wg.y * xp1[of.y]
                 + wg.z * xp1[of.z] + wg.w * xp1[of.w];
        lds_A[buf][p_a][cc_a]     = (__bf16)a0;
        lds_A[buf][p_a][cc_a + 1] = (__bf16)a1;
    };

    v8f acc0 = {}; v8f acc1 = {};

    // ---- prologue: stage chunk 0 into buffer 0 ----
    if (wv == 0) tdm_load_B(wT, addrB0);
    build_A(0, 0);
    if (wv == 0) __builtin_amdgcn_s_wait_tensorcnt(0);
    __syncthreads();

    for (int kc = 0; kc < NCHUNK; ++kc) {
        const int cur = kc & 1, nxt = cur ^ 1;

        // ---- stage chunk kc+1 into the other buffer ----
        if (kc + 1 < NCHUNK) {
            if (wv == 0) tdm_load_B(wT + (kc + 1) * 32, nxt ? addrB1 : addrB0);
            build_A(kc + 1, nxt);
        }

        // ---- WMMA on current buffer: D += A(16x32) * B(32x16) x2 ----
        {
            int row   = lane & 15;
            int khalf = lane >> 4;          // 0: K 0-15, 1: K 16-31
            const v8bf* arow = (const v8bf*)&lds_A[cur][row][0];
            v8bf alo = arow[khalf * 2];
            v8bf ahi = arow[khalf * 2 + 1];
            v16bf A = __builtin_shufflevector(alo, ahi,
                0,1,2,3,4,5,6,7,8,9,10,11,12,13,14,15);

            int col0 = wv * 32 + (lane & 15);
            const v8bf* bp0 = (const v8bf*)&lds_B[cur][col0][khalf * 16];
            v16bf B0 = __builtin_shufflevector(bp0[0], bp0[1],
                0,1,2,3,4,5,6,7,8,9,10,11,12,13,14,15);
            const v8bf* bp1 = (const v8bf*)&lds_B[cur][col0 + 16][khalf * 16];
            v16bf B1 = __builtin_shufflevector(bp1[0], bp1[1],
                0,1,2,3,4,5,6,7,8,9,10,11,12,13,14,15);

            acc0 = __builtin_amdgcn_wmma_f32_16x16x32_bf16(
                       false, A, false, B0, (short)0, acc0, false, false);
            acc1 = __builtin_amdgcn_wmma_f32_16x16x32_bf16(
                       false, A, false, B1, (short)0, acc1, false, false);
        }

        // TDM for nxt must land + everyone must be done reading cur
        if (wv == 0) __builtin_amdgcn_s_wait_tensorcnt(0);
        __syncthreads();
    }

    // ---- store D tiles: VGPR i holds M=i (lanes 0-15) / M=i+8 (lanes 16-31) ----
    int rbase = (lane >> 4) * 8;
    int col   = wv * 32 + (lane & 15);
#pragma unroll
    for (int i = 0; i < 8; ++i) {
        int row = rbase + i;
        y[(size_t)(pix0 + row) * POUT + col]      = acc0[i];
        y[(size_t)(pix0 + row) * POUT + col + 16] = acc1[i];
    }
}

// ---------------------------------------------------------------------
// BatchNorm (training stats) + ReLU
// ---------------------------------------------------------------------
__global__ void k_zero_stats(float* s) {
    if (threadIdx.x < 512) s[threadIdx.x] = 0.f;
}

__global__ void __launch_bounds__(256)
k_bn_stats(const float* __restrict__ y, float* __restrict__ stats) {
    int c  = threadIdx.x;
    int r0 = blockIdx.x * 64;          // 256 blocks * 64 rows = 16384 rows
    float s = 0.f, s2 = 0.f;
    for (int r = 0; r < 64; ++r) {
        float v = y[(size_t)(r0 + r) * POUT + c];
        s += v; s2 += v * v;
    }
    atomicAdd(&stats[c], s);
    atomicAdd(&stats[256 + c], s2);
}

__global__ void k_bn_final(float* stats, const float* __restrict__ gamma,
                           const float* __restrict__ beta) {
    int c = threadIdx.x;
    float inv  = 1.f / (float)(BATCH * HWSZ);
    float mean = stats[c] * inv;
    float var  = stats[256 + c] * inv - mean * mean;
    float sc   = gamma[c] * rsqrtf(var + EPSV);
    stats[512 + c] = sc;
    stats[768 + c] = beta[c] - mean * sc;
}

__global__ void __launch_bounds__(256)
k_bn_apply(const float* __restrict__ y, const float* __restrict__ stats,
           float* __restrict__ out) {
    size_t idx = (size_t)blockIdx.x * 256 + threadIdx.x;   // over B*P*HW
    int b   = (int)(idx >> 20);                            // P*HW = 1<<20
    int rem = (int)(idx & ((1u << 20) - 1));
    int p = rem >> 12, hw = rem & 4095;
    float v = y[((size_t)(b * HWSZ + hw)) * POUT + p];
    v = v * stats[512 + p] + stats[768 + p];
    out[idx] = fmaxf(v, 0.f);
}

// ---------------------------------------------------------------------
extern "C" void kernel_launch(void* const* d_in, const int* in_sizes, int n_in,
                              void* d_out, int out_size, void* d_ws, size_t ws_size,
                              hipStream_t stream) {
    (void)in_sizes; (void)n_in; (void)out_size; (void)ws_size;
    const float* x     = (const float*)d_in[0];
    const float* w_p   = (const float*)d_in[1];
    const float* b_p   = (const float*)d_in[2];
    const float* w_m   = (const float*)d_in[3];
    const float* b_m   = (const float*)d_in[4];
    const float* w     = (const float*)d_in[5];
    const float* gamma = (const float*)d_in[6];
    const float* beta  = (const float*)d_in[7];
    float* out = (float*)d_out;

    char* ws = (char*)d_ws;
    __bf16* wT   = (__bf16*)ws;                                  // 1,179,648 B
    float*  om   = (float*)(ws + 1179648);                       // 1,769,472 B
    float*  yb   = (float*)(ws + 1179648 + 1769472);             // 16,777,216 B
    float*  stats= (float*)(ws + 1179648 + 1769472 + 16777216);  // 4 KB

    k_convert_w  <<<(POUT * KTOT + 255) / 256, 256, 0, stream>>>(w, wT);
    k_offset_mask<<<BATCH * HWSZ,              256, 0, stream>>>(x, w_p, b_p, w_m, b_m, om);
    k_deform_wmma<<<(BATCH * HWSZ) / 16,       256, 0, stream>>>(x, om, wT, yb);
    k_zero_stats <<<1,                         512, 0, stream>>>(stats);
    k_bn_stats   <<<256,                       256, 0, stream>>>(yb, stats);
    k_bn_final   <<<1,                         256, 0, stream>>>(stats, gamma, beta);
    k_bn_apply   <<<(BATCH * POUT * HWSZ) / 256, 256, 0, stream>>>(yb, stats, out);
}
